// BlockInvariantPointAttention_2430951489761
// MI455X (gfx1250) — compile-verified, matrix operand order checked by reading, not run on hardware
//
#include <hip/hip_runtime.h>
#include <hip/hip_bf16.h>
#include <math.h>

// ---------------------------------------------------------------------------
// Block Invariant Point Attention for MI455X (gfx1250, wave32, bf16 WMMA)
// ---------------------------------------------------------------------------
typedef __bf16 bf16_t;
typedef bf16_t v16bf __attribute__((ext_vector_type(16)));
typedef bf16_t bf16x8 __attribute__((ext_vector_type(8)));
typedef float  f32x8 __attribute__((ext_vector_type(8)));
typedef unsigned int u32x4 __attribute__((ext_vector_type(4)));
typedef int i32x4 __attribute__((ext_vector_type(4)));
typedef int i32x8 __attribute__((ext_vector_type(8)));
#define AS1 __attribute__((address_space(1)))
#define AS3 __attribute__((address_space(3)))

constexpr int kN   = 4096;
constexpr int kCS  = 512;
constexpr int kCZ  = 128;
constexpr int kCH  = 64;
constexpr int kH   = 16;
constexpr int kPQ  = 4;
constexpr int kPV  = 8;
constexpr int kBQ  = 32;
constexpr int kBK  = 128;
constexpr int kNB  = 128;
constexpr int kCZ4 = 32;
constexpr int kPW  = 3840;               // concat proj width: q|k|v|qp|kvp
constexpr int kOQ = 0, kOK = 1024, kOV = 2048, kOQP = 3072, kOKVP = 3264;
constexpr int kCAT = 2048;               // H*(CZ4+CH+PV*4)
constexpr float kINF = 100000.0f;

#define WMMA_BF16(a, b, c) \
  __builtin_amdgcn_wmma_f32_16x16x32_bf16(false, (a), false, (b), (short)0, (c), false, false)

// A/B fragment (16x32 bf16): lane<16 -> row m=lane, K = {0..7,16..23};
// lane>=16 -> row m=lane-16, K = {8..15,24..31}. B uses same layout with
// "row" = output column (operand stored N-major, K contiguous).
__device__ __forceinline__ v16bf load_frag_bf16(const bf16_t* base, int ld,
                                                int r0, int c0, int lane) {
  int m = lane & 15;
  int kb = (lane >> 4) << 3;
  const bf16_t* p = base + (size_t)(r0 + m) * ld + c0 + kb;
  bf16x8 lo = *(const bf16x8*)p;
  bf16x8 hi = *(const bf16x8*)(p + 16);
  v16bf f;
#pragma unroll
  for (int i = 0; i < 8; ++i) { f[i] = lo[i]; f[8 + i] = hi[i]; }
  return f;
}

__device__ __forceinline__ v16bf load_frag_f32(const float* base, size_t ld,
                                               int r0, int c0, int lane) {
  int m = lane & 15;
  int kb = (lane >> 4) << 3;
  const float* p = base + (size_t)(r0 + m) * ld + c0 + kb;
  v16bf f;
#pragma unroll
  for (int i = 0; i < 8; ++i) { f[i] = (bf16_t)p[i]; f[8 + i] = (bf16_t)p[16 + i]; }
  return f;
}

// C/D tile: VGPR r, lane -> D[r + 8*(lane>>4)][lane&15]
__device__ __forceinline__ void store_tile(float* base, size_t ld, int r0,
                                           int c0, int lane, f32x8 acc) {
  int n = lane & 15;
  int mh = (lane >> 4) << 3;
#pragma unroll
  for (int r = 0; r < 8; ++r) base[(size_t)(r0 + mh + r) * ld + c0 + n] = acc[r];
}

// ---------------------------------------------------------------------------
// 1) Transpose/convert all weights to bf16, N-major (K contiguous per row)
// ---------------------------------------------------------------------------
__global__ void __launch_bounds__(256)
k_prep(const float* __restrict__ Wq, const float* __restrict__ Wk,
       const float* __restrict__ Wv, const float* __restrict__ Wqp,
       const float* __restrict__ Wkvp, const float* __restrict__ Wb,
       const float* __restrict__ Wdz, const float* __restrict__ Wout,
       bf16_t* __restrict__ wcatT, bf16_t* __restrict__ wbzT,
       bf16_t* __restrict__ woutT) {
  const long total1 = (long)kPW * kCS;
  const long total2 = 48L * kCZ;
  const long total3 = (long)kCS * kCAT;
  const long total = total1 + total2 + total3;
  for (long i = (long)blockIdx.x * blockDim.x + threadIdx.x; i < total;
       i += (long)gridDim.x * blockDim.x) {
    if (i < total1) {
      int col = (int)(i / kCS), cs = (int)(i % kCS);
      float v;
      if (col < kOK)        v = Wq[(size_t)cs * 1024 + col];
      else if (col < kOV)   v = Wk[(size_t)cs * 1024 + (col - kOK)];
      else if (col < kOQP)  v = Wv[(size_t)cs * 1024 + (col - kOV)];
      else if (col < kOKVP) v = Wqp[(size_t)cs * 192 + (col - kOQP)];
      else                  v = Wkvp[(size_t)cs * 576 + (col - kOKVP)];
      wcatT[i] = (bf16_t)v;
    } else if (i < total1 + total2) {
      long j = i - total1;
      int col = (int)(j / kCZ), cz = (int)(j % kCZ);
      float v = (col < kH) ? Wb[(size_t)cz * kH + col]
                           : Wdz[(size_t)cz * kCZ4 + (col - kH)];
      wbzT[j] = (bf16_t)v;
    } else {
      long j = i - total1 - total2;
      int col = (int)(j / kCAT), k = (int)(j % kCAT);
      woutT[j] = (bf16_t)Wout[(size_t)k * kCS + col];
    }
  }
}

// ---------------------------------------------------------------------------
// 2) LayerNorm of s -> bf16 (one block per row of 512)
// ---------------------------------------------------------------------------
__global__ void __launch_bounds__(256)
k_ln_s(const float* __restrict__ s, const float* __restrict__ g,
       const float* __restrict__ b, bf16_t* __restrict__ sn) {
  __shared__ float red[8];
  __shared__ float stat[2];
  const int row = blockIdx.x, t = threadIdx.x;
  const float* x = s + (size_t)row * kCS;
  float v0 = x[t], v1 = x[t + 256];
  float sum = v0 + v1;
#pragma unroll
  for (int o = 16; o > 0; o >>= 1) sum += __shfl_xor(sum, o, 32);
  if ((t & 31) == 0) red[t >> 5] = sum;
  __syncthreads();
  if (t == 0) {
    float a = 0.f;
    for (int i = 0; i < 8; ++i) a += red[i];
    stat[0] = a * (1.0f / kCS);
  }
  __syncthreads();
  float mean = stat[0];
  float d0 = v0 - mean, d1 = v1 - mean;
  float var = d0 * d0 + d1 * d1;
#pragma unroll
  for (int o = 16; o > 0; o >>= 1) var += __shfl_xor(var, o, 32);
  __syncthreads();
  if ((t & 31) == 0) red[t >> 5] = var;
  __syncthreads();
  if (t == 0) {
    float a = 0.f;
    for (int i = 0; i < 8; ++i) a += red[i];
    stat[1] = rsqrtf(a * (1.0f / kCS) + 1e-5f);
  }
  __syncthreads();
  float rstd = stat[1];
  sn[(size_t)row * kCS + t]       = (bf16_t)(d0 * rstd * g[t] + b[t]);
  sn[(size_t)row * kCS + t + 256] = (bf16_t)(d1 * rstd * g[t + 256] + b[t + 256]);
}

// ---------------------------------------------------------------------------
// 3) All projections fused: P = s_n @ [Wq|Wk|Wv|Wqp|Wkvp]  (4096x512x3840)
//    2x4 register-blocked: 8 WMMAs per 6 fragment loads per k-step.
// ---------------------------------------------------------------------------
__global__ void __launch_bounds__(256)
k_proj(const bf16_t* __restrict__ sn, const bf16_t* __restrict__ wT,
       float* __restrict__ P) {
  const int lane = threadIdx.x & 31, w = threadIdx.x >> 5;
  const int gid = blockIdx.x * 8 + w;
  const int ngrp = (kPW / 16) / 4;          // 60 column groups
  const int mt = (gid / ngrp) * 2;
  const int nt = (gid % ngrp) * 4;
  f32x8 acc[2][4] = {};
#pragma unroll 2
  for (int k0 = 0; k0 < kCS; k0 += 32) {
    v16bf a0 = load_frag_bf16(sn, kCS, (mt + 0) * 16, k0, lane);
    v16bf a1 = load_frag_bf16(sn, kCS, (mt + 1) * 16, k0, lane);
#pragma unroll
    for (int j = 0; j < 4; ++j) {
      v16bf bv = load_frag_bf16(wT, kCS, (nt + j) * 16, k0, lane);
      acc[0][j] = WMMA_BF16(a0, bv, acc[0][j]);
      acc[1][j] = WMMA_BF16(a1, bv, acc[1][j]);
    }
  }
#pragma unroll
  for (int i = 0; i < 2; ++i)
#pragma unroll
    for (int j = 0; j < 4; ++j)
      store_tile(P, kPW, (mt + i) * 16, (nt + j) * 16, lane, acc[i][j]);
}

// ---------------------------------------------------------------------------
// 4) Per-node frame transform of raw q/kv points: R@p + t
// ---------------------------------------------------------------------------
__global__ void __launch_bounds__(256)
k_points(const float* __restrict__ P, const float* __restrict__ rots,
         const float* __restrict__ trans, float* __restrict__ qpts,
         float* __restrict__ kvpts) {
  const int node = blockIdx.x, t = threadIdx.x;
  const float* R = rots + (size_t)node * 9;
  const float tx = trans[node * 3 + 0];
  const float ty = trans[node * 3 + 1];
  const float tz = trans[node * 3 + 2];
  const float* Prow = P + (size_t)node * kPW;
  if (t < 64) {
    const float* raw = Prow + kOQP + t * 3;
    float x = raw[0], y = raw[1], zc = raw[2];
    float* o = qpts + ((size_t)node * 64 + t) * 3;
    o[0] = R[0] * x + R[1] * y + R[2] * zc + tx;
    o[1] = R[3] * x + R[4] * y + R[5] * zc + ty;
    o[2] = R[6] * x + R[7] * y + R[8] * zc + tz;
  } else {
    int hp = t - 64;  // 0..191
    const float* raw = Prow + kOKVP + hp * 3;
    float x = raw[0], y = raw[1], zc = raw[2];
    float* o = kvpts + ((size_t)node * 192 + hp) * 3;
    o[0] = R[0] * x + R[1] * y + R[2] * zc + tx;
    o[1] = R[3] * x + R[4] * y + R[5] * zc + ty;
    o[2] = R[6] * x + R[7] * y + R[8] * zc + tz;
  }
}

// ---------------------------------------------------------------------------
// 5) Fused z pipeline: TDM/async-stage z tile to LDS, LN, then
//    [bias|pair_z] = z_n @ [Wb|Wdz] via WMMA (128x128x48 per tile).
//    Single pass over the 268 MB z tensor — the roofline-critical stream.
// ---------------------------------------------------------------------------
__global__ void __launch_bounds__(256)
k_zfuse(const float* __restrict__ z, const float* __restrict__ gz,
        const float* __restrict__ bz, const bf16_t* __restrict__ wbzT,
        float* __restrict__ biasb, float* __restrict__ pairzT) {
  const int blk = blockIdx.x;  // nb*32 + q
  const int nb = blk >> 5, q = blk & 31;
  const int t = threadIdx.x, lane = t & 31, w = t >> 5;
  __shared__ __align__(16) float  zf[64][kCZ];   // 32 KB raw staging (half tile)
  __shared__ __align__(16) bf16_t zn[kBK][kCZ];  // 32 KB LN'd tile
  const float* zsrc = z + (size_t)blk * kBK * kCZ;

  for (int half = 0; half < 2; ++half) {
    const float* gsrc = zsrc + (size_t)half * 64 * kCZ;
#if defined(__gfx1250__) && __has_builtin(__builtin_amdgcn_tensor_load_to_lds)
    // Tensor Data Mover: one 1-D descriptor per 32 KB half-tile, wave 0 only.
    // clang-23 form: (u32x4 g0, i32x8 g1, i32x4 g2, i32x4 g3, i32x8 g4, i32 cpol)
    if (w == 0) {
      unsigned long long ga = (unsigned long long)(uintptr_t)gsrc;
      unsigned lds = (unsigned)(uintptr_t)(void*)&zf[0][0];
      u32x4 g0;
      i32x8 g1;
      i32x4 g2 = {0, 0, 0, 0}, g3 = {0, 0, 0, 0};
      i32x8 g4 = {0, 0, 0, 0, 0, 0, 0, 0};
      g0[0] = 1u;                                          // D# count = 1
      g0[1] = lds;                                         // lds_addr
      g0[2] = (unsigned)(ga & 0xffffffffu);                // global_addr[31:0]
      g0[3] = (unsigned)((ga >> 32) & 0x01ffffffu) | (2u << 30);  // addr[56:32]|type=2
      g1[0] = (int)(2u << 16);   // workgroup_mask=0, data_size=4B
      g1[1] = (int)(8192u << 16);// tensor_dim0 = 8192 elems (lo16 @ bits63:48)
      g1[2] = (int)(1u << 16);   // tensor_dim0 hi16 = 0; tensor_dim1 = 1
      g1[3] = (int)(8192u << 16);// tensor_dim1 hi16 = 0; tile_dim0 = 8192
      g1[4] = 1;                 // tile_dim1 = 1
      g1[5] = 8192;              // tensor_dim0_stride
      g1[6] = 0;
      g1[7] = 0;
      __builtin_amdgcn_tensor_load_to_lds(g0, g1, g2, g3, g4, 0);
      __builtin_amdgcn_s_wait_tensorcnt(0);
    }
    __syncthreads();
#elif defined(__gfx1250__) && __has_builtin(__builtin_amdgcn_global_load_async_to_lds_b128)
#pragma unroll
    for (int it = 0; it < 8; ++it) {
      int e = it * 256 + t;  // 16-byte chunk id (2048 chunks = 32 KB)
      __builtin_amdgcn_global_load_async_to_lds_b128(
          (AS1 void*)(gsrc + e * 4), (AS3 void*)(&zf[0][0] + e * 4), 0, 0);
    }
    asm volatile("s_wait_asynccnt 0" ::: "memory");
    __syncthreads();
#else
    for (int e = t; e < 64 * kCZ / 4; e += 256)
      ((float4*)&zf[0][0])[e] = ((const float4*)gsrc)[e];
    __syncthreads();
#endif
    // LayerNorm from LDS: 4 lanes per row, 64 rows per half
    {
      int rl = t >> 2, qtr = t & 3;
      const float* x = &zf[rl][qtr * 32];
      float sum = 0.f;
#pragma unroll
      for (int i = 0; i < 32; ++i) sum += x[i];
      sum += __shfl_xor(sum, 1, 32);
      sum += __shfl_xor(sum, 2, 32);
      float mean = sum * (1.0f / kCZ);
      float var = 0.f;
#pragma unroll
      for (int i = 0; i < 32; ++i) { float d = x[i] - mean; var += d * d; }
      var += __shfl_xor(var, 1, 32);
      var += __shfl_xor(var, 2, 32);
      float rstd = rsqrtf(var * (1.0f / kCZ) + 1e-5f);
      int row = half * 64 + rl;
#pragma unroll
      for (int i = 0; i < 32; ++i) {
        int c = qtr * 32 + i;
        zn[row][c] = (bf16_t)((x[i] - mean) * rstd * gz[c] + bz[c]);
      }
    }
    __syncthreads();  // zf fully consumed before next half overwrites it
  }

  f32x8 a0 = {}, a1 = {}, a2 = {};
#pragma unroll
  for (int k0 = 0; k0 < kCZ; k0 += 32) {
    v16bf av = load_frag_bf16(&zn[0][0], kCZ, w * 16, k0, lane);
    v16bf b0 = load_frag_bf16(wbzT, kCZ, 0, k0, lane);
    v16bf b1 = load_frag_bf16(wbzT, kCZ, 16, k0, lane);
    v16bf b2 = load_frag_bf16(wbzT, kCZ, 32, k0, lane);
    a0 = WMMA_BF16(av, b0, a0);
    a1 = WMMA_BF16(av, b1, a1);
    a2 = WMMA_BF16(av, b2, a2);
  }
  int nloc = lane & 15, mh = (lane >> 4) << 3;
#pragma unroll
  for (int r = 0; r < 8; ++r) {
    int key = w * 16 + mh + r;
    biasb[(((size_t)nb * kH + nloc) * kBQ + q) * kBK + key] = a0[r];
    pairzT[((size_t)blk * kCZ4 + nloc) * kBK + key]         = a1[r];
    pairzT[((size_t)blk * kCZ4 + 16 + nloc) * kBK + key]    = a2[r];
  }
}

// ---------------------------------------------------------------------------
// 6) Attention core. One block per (nb, h). Extended 96-dim Q/K vectors fold
//    the point cross-term into the score WMMA; softmax; o and o_pt WMMAs.
//    48 KB LDS arena reused across phases.
// ---------------------------------------------------------------------------
__global__ void __launch_bounds__(256)
k_attn(const float* __restrict__ P, const float* __restrict__ qpts,
       const float* __restrict__ kvpts, const float* __restrict__ biasb,
       const int* __restrict__ key_idx, const float* __restrict__ s_mask,
       const float* __restrict__ headw, const float* __restrict__ trans,
       const float* __restrict__ rots, float* __restrict__ pattn,
       float* __restrict__ Cc) {
  const int n = blockIdx.x >> 4;
  const int h = blockIdx.x & 15;
  const int t = threadIdx.x, lane = t & 31, w = t >> 5;

  __shared__ __align__(16) char smem[49152];
  bf16_t (*qe)[96]   = (bf16_t(*)[96])(smem);            // phase A: 6 KB
  bf16_t (*ke)[96]   = (bf16_t(*)[96])(smem + 8192);     // phase A: 24 KB
  float  (*S)[kBK]   = (float(*)[kBK])(smem + 32768);    // 16 KB
  bf16_t (*ab)[kBK]  = (bf16_t(*)[kBK])(smem);           // phase B over qe
  bf16_t (*vbT)[kBK] = (bf16_t(*)[kBK])(smem + 8192);    // phase B over ke
  bf16_t (*vpT)[kBK] = (bf16_t(*)[kBK])(smem + 24576);
  float  (*opt)[32]  = (float(*)[32])(smem + 32768);     // phase C over S
  __shared__ float rowterm[kBQ], qmv[kBQ], colterm[kBK], kmv[kBK];

  const float hw = log1pf(__expf(headw[h])) * 0.1360827635f;  // sqrt(1/54)
  const float c1 = 0.0721687836f;                             // sqrt(1/(3*CH))

  // ---- phase 1: stage extended Q and gathered extended K ----
  for (int e = t; e < kBQ * 96; e += 256) {
    int qi = e / 96, d = e - qi * 96;
    int node = n * kBQ + qi;
    float v = 0.f;
    if (d < 64) v = P[(size_t)node * kPW + kOQ + h * kCH + d] * c1;
    else if (d < 76) {
      int j = d - 64;
      v = qpts[((size_t)node * 64 + h * kPQ + j / 3) * 3 + (j % 3)] * hw;
    }
    qe[qi][d] = (bf16_t)v;
  }
  for (int e = t; e < kBK * 96; e += 256) {
    int ki = e / 96, d = e - ki * 96;
    int raw = key_idx[n * kBK + ki];
    bool valid = (raw >= 0) && (raw < kN);
    int node = raw < 0 ? 0 : (raw >= kN ? kN - 1 : raw);
    float vf = valid ? 1.f : 0.f;
    float v = 0.f;
    if (d < 64) v = P[(size_t)node * kPW + kOK + h * kCH + d] * vf;
    else if (d < 76) {
      int j = d - 64;
      v = kvpts[((size_t)node * 192 + h * 12 + j / 3) * 3 + (j % 3)] * vf;
    }
    ke[ki][d] = (bf16_t)v;
  }
  if (t < kBQ) {
    int node = n * kBQ + t;
    float ss = 0.f;
    for (int j = 0; j < 12; ++j) {
      float c = qpts[((size_t)node * 64 + h * kPQ + j / 3) * 3 + (j % 3)];
      ss += c * c;
    }
    rowterm[t] = -0.5f * hw * ss;
    qmv[t] = s_mask[node];
  }
  if (t < kBK) {
    int raw = key_idx[n * kBK + t];
    bool valid = (raw >= 0) && (raw < kN);
    int node = raw < 0 ? 0 : (raw >= kN ? kN - 1 : raw);
    float vf = valid ? 1.f : 0.f;
    float ss = 0.f;
    for (int j = 0; j < 12; ++j) {
      float c = kvpts[((size_t)node * 192 + h * 12 + j / 3) * 3 + (j % 3)] * vf;
      ss += c * c;
    }
    colterm[t] = -0.5f * hw * ss;
    kmv[t] = s_mask[node] * vf;
  }
  __syncthreads();

  // ---- phase 2: S = qe @ ke^T (2x8 tiles, K=96) ----
  for (int tt = w; tt < 16; tt += 8) {
    int mt = tt & 1, nt = tt >> 1;
    f32x8 acc = {};
#pragma unroll
    for (int k0 = 0; k0 < 96; k0 += 32) {
      v16bf av = load_frag_bf16(&qe[0][0], 96, mt * 16, k0, lane);
      v16bf bv = load_frag_bf16(&ke[0][0], 96, nt * 16, k0, lane);
      acc = WMMA_BF16(av, bv, acc);
    }
    store_tile(&S[0][0], kBK, mt * 16, nt * 16, lane, acc);
  }
  __syncthreads();

  // ---- phase 3a: bias/mask epilogue + stage vbT/vpT over dead ke ----
  for (int e = t; e < kBQ * kBK; e += 256) {
    int qi = e >> 7, ki = e & 127;
    S[qi][ki] += 0.5773502692f * biasb[(((size_t)n * kH + h) * kBQ + qi) * kBK + ki] +
                 rowterm[qi] + colterm[ki] + kINF * (qmv[qi] * kmv[ki] - 1.0f);
  }
  for (int e = t; e < kBK * kCH; e += 256) {
    int ki = e >> 6, d = e & 63;
    int raw = key_idx[n * kBK + ki];
    bool valid = (raw >= 0) && (raw < kN);
    int node = raw < 0 ? 0 : (raw >= kN ? kN - 1 : raw);
    float vf = valid ? 1.f : 0.f;
    vbT[d][ki] = (bf16_t)(P[(size_t)node * kPW + kOV + h * kCH + d] * vf);
  }
  for (int e = t; e < kBK * 32; e += 256) {
    int ki = e >> 5, d = e & 31;
    int raw = key_idx[n * kBK + ki];
    bool valid = (raw >= 0) && (raw < kN);
    int node = raw < 0 ? 0 : (raw >= kN ? kN - 1 : raw);
    float vf = valid ? 1.f : 0.f;
    float v = 0.f;
    if (d < 24)
      v = kvpts[((size_t)node * 192 + h * 12 + kPQ + d / 3) * 3 + (d % 3)] * vf;
    vpT[d][ki] = (bf16_t)v;
  }
  __syncthreads();

  // ---- phase 3b: softmax over keys (8 lanes per row), emit probs ----
  {
    int qi = t >> 3, j = t & 7;
    float mx = -3.0e38f;
    for (int ki = j; ki < kBK; ki += 8) mx = fmaxf(mx, S[qi][ki]);
#pragma unroll
    for (int o = 4; o > 0; o >>= 1) mx = fmaxf(mx, __shfl_xor(mx, o, 32));
    float sum = 0.f;
    for (int ki = j; ki < kBK; ki += 8) {
      float ev = __expf(S[qi][ki] - mx);
      S[qi][ki] = ev;
      sum += ev;
    }
#pragma unroll
    for (int o = 4; o > 0; o >>= 1) sum += __shfl_xor(sum, o, 32);
    float inv = 1.0f / sum;
    for (int ki = j; ki < kBK; ki += 8) {
      float p = S[qi][ki] * inv;
      ab[qi][ki] = (bf16_t)p;
      pattn[(((size_t)n * kH + h) * kBQ + qi) * kBK + ki] = p;
    }
  }
  __syncthreads();

  // ---- phase 4: o = a@v (8 tiles), o_pt = a@v_pts (4 tiles) ----
  {
    int mt = w & 1, nt = w >> 1;
    f32x8 acc = {};
#pragma unroll
    for (int k0 = 0; k0 < kBK; k0 += 32) {
      v16bf av = load_frag_bf16(&ab[0][0], kBK, mt * 16, k0, lane);
      v16bf bv = load_frag_bf16(&vbT[0][0], kBK, nt * 16, k0, lane);
      acc = WMMA_BF16(av, bv, acc);
    }
    int nloc = lane & 15, mh = (lane >> 4) << 3;
#pragma unroll
    for (int r = 0; r < 8; ++r) {
      int qi = mt * 16 + mh + r;
      Cc[(size_t)(n * kBQ + qi) * kCAT + h * kCH + nt * 16 + nloc] = acc[r];
    }
  }
  if (w < 4) {
    int mt = w & 1, nt = w >> 1;
    f32x8 acc = {};
#pragma unroll
    for (int k0 = 0; k0 < kBK; k0 += 32) {
      v16bf av = load_frag_bf16(&ab[0][0], kBK, mt * 16, k0, lane);
      v16bf bv = load_frag_bf16(&vpT[0][0], kBK, nt * 16, k0, lane);
      acc = WMMA_BF16(av, bv, acc);
    }
    store_tile(&opt[0][0], 32, mt * 16, nt * 16, lane, acc);
  }
  __syncthreads();

  // ---- phase 5: inverse frame transform of o_pt + norms ----
  {
    int qi = t >> 3, p = t & 7;
    int node = n * kBQ + qi;
    const float* R = rots + (size_t)node * 9;
    float dx = opt[qi][p * 3 + 0] - trans[node * 3 + 0];
    float dy = opt[qi][p * 3 + 1] - trans[node * 3 + 1];
    float dz = opt[qi][p * 3 + 2] - trans[node * 3 + 2];
    float rx = R[0] * dx + R[3] * dy + R[6] * dz;  // R^T d
    float ry = R[1] * dx + R[4] * dy + R[7] * dz;
    float rz = R[2] * dx + R[5] * dy + R[8] * dz;
    size_t row = (size_t)node * kCAT;
    Cc[row + 1024 + (h * kPV + p) * 3 + 0] = rx;
    Cc[row + 1024 + (h * kPV + p) * 3 + 1] = ry;
    Cc[row + 1024 + (h * kPV + p) * 3 + 2] = rz;
    Cc[row + 1408 + h * kPV + p] = sqrtf(rx * rx + ry * ry + rz * rz + 1e-8f);
  }
}

// ---------------------------------------------------------------------------
// 7) o_pair = a(h,k) @ pair_z(k,c) per (nb,q). 16x128x32 WMMA, 2 waves/block.
// ---------------------------------------------------------------------------
__global__ void __launch_bounds__(64)
k_opair(const float* __restrict__ pattn, const float* __restrict__ pairzT,
        float* __restrict__ Cc) {
  const int blk = blockIdx.x;  // == node == nb*32 + q
  const int nb = blk >> 5, q = blk & 31;
  const int lane = threadIdx.x & 31, w = threadIdx.x >> 5;
  const float* A = pattn + ((size_t)nb * kH * kBQ + q) * kBK;  // h-stride BQ*BK
  const float* Bz = pairzT + (size_t)blk * kCZ4 * kBK;
  f32x8 acc = {};
#pragma unroll
  for (int k0 = 0; k0 < kBK; k0 += 32) {
    v16bf av = load_frag_f32(A, (size_t)kBQ * kBK, 0, k0, lane);
    v16bf bv = load_frag_f32(Bz, kBK, w * 16, k0, lane);
    acc = WMMA_BF16(av, bv, acc);
  }
  int nloc = lane & 15, mh = (lane >> 4) << 3;
#pragma unroll
  for (int r = 0; r < 8; ++r) {
    int hh = mh + r;
    Cc[(size_t)blk * kCAT + 1536 + hh * kCZ4 + w * 16 + nloc] = acc[r];
  }
}

// ---------------------------------------------------------------------------
// 8) out = concat @ Wout   (4096x2048x512), 2x4 register-blocked
// ---------------------------------------------------------------------------
__global__ void __launch_bounds__(256)
k_out(const float* __restrict__ Cc, const bf16_t* __restrict__ woutT,
      float* __restrict__ out) {
  const int lane = threadIdx.x & 31, w = threadIdx.x >> 5;
  const int gid = blockIdx.x * 8 + w;
  const int ngrp = (kCS / 16) / 4;          // 8 column groups
  const int mt = (gid / ngrp) * 2;
  const int nt = (gid % ngrp) * 4;
  f32x8 acc[2][4] = {};
  for (int k0 = 0; k0 < kCAT; k0 += 32) {
    v16bf a0 = load_frag_f32(Cc, kCAT, (mt + 0) * 16, k0, lane);
    v16bf a1 = load_frag_f32(Cc, kCAT, (mt + 1) * 16, k0, lane);
#pragma unroll
    for (int j = 0; j < 4; ++j) {
      v16bf bv = load_frag_bf16(woutT, kCAT, (nt + j) * 16, k0, lane);
      acc[0][j] = WMMA_BF16(a0, bv, acc[0][j]);
      acc[1][j] = WMMA_BF16(a1, bv, acc[1][j]);
    }
  }
#pragma unroll
  for (int i = 0; i < 2; ++i)
#pragma unroll
    for (int j = 0; j < 4; ++j)
      store_tile(out, kCS, (mt + i) * 16, (nt + j) * 16, lane, acc[i][j]);
}

// ---------------------------------------------------------------------------
extern "C" void kernel_launch(void* const* d_in, const int* in_sizes, int n_in,
                              void* d_out, int out_size, void* d_ws,
                              size_t ws_size, hipStream_t stream) {
  (void)in_sizes; (void)n_in; (void)out_size; (void)ws_size;
  const float* s       = (const float*)d_in[0];
  const float* z       = (const float*)d_in[1];
  const float* trans   = (const float*)d_in[2];
  const float* rots    = (const float*)d_in[3];
  const float* s_mask  = (const float*)d_in[4];
  const int*   key_idx = (const int*)d_in[5];
  const float* Wq      = (const float*)d_in[6];
  const float* Wk      = (const float*)d_in[7];
  const float* Wv      = (const float*)d_in[8];
  const float* Wqp     = (const float*)d_in[9];
  const float* Wkvp    = (const float*)d_in[10];
  const float* Wb      = (const float*)d_in[11];
  const float* Wdz     = (const float*)d_in[12];
  const float* headw   = (const float*)d_in[13];
  const float* Wout    = (const float*)d_in[14];
  const float* g_s     = (const float*)d_in[15];
  const float* b_s     = (const float*)d_in[16];
  const float* g_z     = (const float*)d_in[17];
  const float* b_z     = (const float*)d_in[18];
  float* out = (float*)d_out;

  char* ws = (char*)d_ws;
  size_t off = 0;
  auto alloc = [&](size_t bytes) -> void* {
    void* p = ws + off;
    off += (bytes + 255) & ~(size_t)255;
    return p;
  };
  bf16_t* snb    = (bf16_t*)alloc((size_t)kN * kCS * 2);
  bf16_t* wcatT  = (bf16_t*)alloc((size_t)kPW * kCS * 2);
  bf16_t* wbzT   = (bf16_t*)alloc((size_t)48 * kCZ * 2);
  bf16_t* woutT  = (bf16_t*)alloc((size_t)kCAT * kCS * 2);
  float*  P      = (float*)alloc((size_t)kN * kPW * 4);
  float*  qpts   = (float*)alloc((size_t)kN * 64 * 3 * 4);
  float*  kvpts  = (float*)alloc((size_t)kN * 192 * 3 * 4);
  float*  biasb  = (float*)alloc((size_t)kNB * kH * kBQ * kBK * 4);
  float*  pairzT = (float*)alloc((size_t)kNB * kBQ * kCZ4 * kBK * 4);
  float*  pattn  = (float*)alloc((size_t)kNB * kH * kBQ * kBK * 4);
  float*  Cc     = (float*)alloc((size_t)kN * kCAT * 4);

  k_prep<<<2048, 256, 0, stream>>>(Wq, Wk, Wv, Wqp, Wkvp, Wb, Wdz, Wout,
                                   wcatT, wbzT, woutT);
  k_ln_s<<<kN, 256, 0, stream>>>(s, g_s, b_s, snb);
  k_proj<<<((kN / 16) / 2) * ((kPW / 16) / 4) / 8, 256, 0, stream>>>(snb, wcatT, P);
  k_points<<<kN, 256, 0, stream>>>(P, rots, trans, qpts, kvpts);
  k_zfuse<<<kNB * kBQ, 256, 0, stream>>>(z, g_z, b_z, wbzT, biasb, pairzT);
  k_attn<<<kNB * kH, 256, 0, stream>>>(P, qpts, kvpts, biasb, key_idx, s_mask,
                                       headw, trans, rots, pattn, Cc);
  k_opair<<<kNB * kBQ, 64, 0, stream>>>(pattn, pairzT, Cc);
  k_out<<<((kN / 16) / 2) * ((kCS / 16) / 4) / 8, 256, 0, stream>>>(Cc, woutT, out);
}
